// MultiTaskClassifier_22419729285374
// MI455X (gfx1250) — compile-verified, matrix operand order checked by reading, not run on hardware
//
#include <hip/hip_runtime.h>

// MultiTaskClassifier: out[pos*256 + b] = b2[k] + sum_o w2[k][o]*relu(b1[k][o] + emb[b,:]·W1[k][o,:])
// k = task_ids[pos]. bf16 WMMA (f32 accum), 8 tasks per workgroup,
// W1 staged via double-buffered LDS, embeddings pre-converted to bf16 in d_ws.

#define NUM_B     256
#define HID       1024
#define TPW       8      // tasks per workgroup
#define NTILES    4      // 16-wide N tiles (TPW*8/16)
#define KSTEP     32
#define SB_STRIDE 40     // bf16 halves per staged B row (32 data + 8 pad -> 80B, bank-conflict-free)

typedef __attribute__((ext_vector_type(16))) __bf16 v16bf;
typedef __attribute__((ext_vector_type(8)))  __bf16 v8bf;
typedef __attribute__((ext_vector_type(8)))  float  v8f;
typedef __attribute__((ext_vector_type(4)))  float  v4f;

// ---- pre-pass: convert shared embeddings f32 -> bf16 once (saves per-WG cvt + halves A bandwidth)
__global__ __launch_bounds__(256)
void emb_to_bf16_kernel(const float* __restrict__ emb, __bf16* __restrict__ dst) {
    int i = (blockIdx.x * 256 + threadIdx.x) * 8;   // 128 blocks cover 256*1024 elements
    v4f a = *(const v4f*)(emb + i);
    v4f b = *(const v4f*)(emb + i + 4);
    v8bf y;
    #pragma unroll
    for (int j = 0; j < 4; ++j) {
        y[j]     = (__bf16)a[j];
        y[4 + j] = (__bf16)b[j];
    }
    *(v8bf*)(dst + i) = y;
}

template <bool PRE>
__global__ __launch_bounds__(256, 1)
void mtc_wmma_kernel(const float*  __restrict__ emb,    // [256, 1024] f32 (used when !PRE)
                     const __bf16* __restrict__ embbf,  // [256, 1024] bf16 (used when PRE)
                     const float*  __restrict__ W1,     // [10000, 8, 1024]
                     const float*  __restrict__ b1,     // [10000, 8]
                     const float*  __restrict__ W2,     // [10000, 8]
                     const float*  __restrict__ b2,     // [10000]
                     const int*    __restrict__ tids,   // [K]
                     float*        __restrict__ out,    // [K*256]
                     int K)
{
    __shared__ __bf16 sB[2][64 * SB_STRIDE];   // double-buffered W1 stage (2 x 5 KB)

    const int tid  = threadIdx.x;
    const int lane = tid & 31;
    const int wave = tid >> 5;      // 0..7 -> 32 M-rows each
    const int lr   = lane & 15;
    const int kg   = lane >> 4;     // K-group within fragment
    const int wg   = blockIdx.x;

    // ---------- staging role: each thread owns B-row sn, 8-wide K chunk ----------
    const int sn = tid >> 2;            // 0..63 : n = task_local*8 + o
    const int so = (tid & 3) * 8;       // 0,8,16,24 within the 32-wide K step
    int spos  = wg * TPW + (sn >> 3);
    int stask = (spos < K) ? tids[spos] : 0;
    const float* w1p = W1 + (size_t)stask * (8 * HID) + (size_t)(sn & 7) * HID + so;
    const int sidx = sn * SB_STRIDE + so;

    // ---------- compute role ----------
    const int mrow = wave * 32;

    v8f zero = {};
    v8f acc[2][NTILES];
    #pragma unroll
    for (int mt = 0; mt < 2; ++mt)
        #pragma unroll
        for (int nt = 0; nt < NTILES; ++nt)
            acc[mt][nt] = zero;

    // stage chunk 0 into buffer 0
    {
        v4f x0 = *(const v4f*)(w1p);
        v4f x1 = *(const v4f*)(w1p + 4);
        v8bf y;
        #pragma unroll
        for (int j = 0; j < 4; ++j) { y[j] = (__bf16)x0[j]; y[4 + j] = (__bf16)x1[j]; }
        *(v8bf*)(&sB[0][sidx]) = y;
    }
    __syncthreads();

    int buf = 0;
    for (int kb = 0; kb < HID; kb += KSTEP) {
        // stage next W1 chunk into the other buffer while current chunk computes
        if (kb + KSTEP < HID) {
            __builtin_prefetch(w1p + kb + 2 * KSTEP, 0, 1);   // global_prefetch for the chunk after next
            v4f x0 = *(const v4f*)(w1p + kb + KSTEP);
            v4f x1 = *(const v4f*)(w1p + kb + KSTEP + 4);
            v8bf y;
            #pragma unroll
            for (int j = 0; j < 4; ++j) { y[j] = (__bf16)x0[j]; y[4 + j] = (__bf16)x1[j]; }
            *(v8bf*)(&sB[buf ^ 1][sidx]) = y;
        }

        // A fragments. 16-bit A layout: halves 0..7 <- K=kg*8.., halves 8..15 <- K=16+kg*8..
        v16bf afrag[2];
        #pragma unroll
        for (int mt = 0; mt < 2; ++mt) {
            const int arow = mrow + mt * 16 + lr;
            v16bf a;
            if (PRE) {
                const __bf16* p = embbf + (size_t)arow * HID + kb + kg * 8;
                v8bf lo = *(const v8bf*)(p);
                v8bf hi = *(const v8bf*)(p + 16);
                #pragma unroll
                for (int j = 0; j < 8; ++j) { a[j] = lo[j]; a[8 + j] = hi[j]; }
            } else {
                const float* p = emb + (size_t)arow * HID + kb + kg * 8;
                v4f f0 = *(const v4f*)(p);
                v4f f1 = *(const v4f*)(p + 4);
                v4f f2 = *(const v4f*)(p + 16);
                v4f f3 = *(const v4f*)(p + 20);
                #pragma unroll
                for (int j = 0; j < 4; ++j) {
                    a[j]      = (__bf16)f0[j];
                    a[4 + j]  = (__bf16)f1[j];
                    a[8 + j]  = (__bf16)f2[j];
                    a[12 + j] = (__bf16)f3[j];
                }
            }
            afrag[mt] = a;
        }

        // B fragments from LDS (col = nt*16 + lr, same K lane-layout rule)
        v16bf bfrag[NTILES];
        #pragma unroll
        for (int nt = 0; nt < NTILES; ++nt) {
            const __bf16* q = &sB[buf][(nt * 16 + lr) * SB_STRIDE + kg * 8];
            v8bf lo = *(const v8bf*)(q);
            v8bf hi = *(const v8bf*)(q + 16);
            v16bf bb;
            #pragma unroll
            for (int j = 0; j < 8; ++j) { bb[j] = lo[j]; bb[8 + j] = hi[j]; }
            bfrag[nt] = bb;
        }

        // 8 WMMAs per K-step per wave
        #pragma unroll
        for (int mt = 0; mt < 2; ++mt)
            #pragma unroll
            for (int nt = 0; nt < NTILES; ++nt)
                acc[mt][nt] = __builtin_amdgcn_wmma_f32_16x16x32_bf16(
                    false, afrag[mt], false, bfrag[nt],
                    (short)0, acc[mt][nt], false, false);

        __syncthreads();   // single barrier per step: next-stage writes went to the other buffer
        buf ^= 1;
    }

    // ---------- epilogue: bias + relu + layer-2 dot over o (8 adjacent lanes) ----------
    const int o  = lane & 7;
    const int tl = (lane >> 3) & 1;   // which task inside the 16-wide N tile

    float vb1[NTILES], vw2[NTILES], vb2[NTILES];
    int   opos[NTILES];
    bool  valid[NTILES];
    #pragma unroll
    for (int nt = 0; nt < NTILES; ++nt) {
        int pos   = wg * TPW + nt * 2 + tl;
        valid[nt] = (pos < K);
        int task  = valid[nt] ? tids[pos] : 0;
        vb1[nt] = b1[(size_t)task * 8 + o];
        vw2[nt] = W2[(size_t)task * 8 + o];
        vb2[nt] = b2[task];
        opos[nt] = pos;
    }

    #pragma unroll
    for (int mt = 0; mt < 2; ++mt) {
        #pragma unroll
        for (int nt = 0; nt < NTILES; ++nt) {
            #pragma unroll
            for (int r = 0; r < 8; ++r) {
                float c = acc[mt][nt][r] + vb1[nt];
                c = fmaxf(c, 0.0f) * vw2[nt];
                c += __shfl_xor(c, 1, 32);
                c += __shfl_xor(c, 2, 32);
                c += __shfl_xor(c, 4, 32);
                if (o == 0 && valid[nt]) {
                    int row = mrow + mt * 16 + (kg ? 8 : 0) + r;
                    out[(size_t)opos[nt] * NUM_B + row] = c + vb2[nt];
                }
            }
        }
    }
}

extern "C" void kernel_launch(void* const* d_in, const int* in_sizes, int n_in,
                              void* d_out, int out_size, void* d_ws, size_t ws_size,
                              hipStream_t stream) {
    const float* emb  = (const float*)d_in[0];
    const float* W1   = (const float*)d_in[1];
    const float* b1   = (const float*)d_in[2];
    const float* W2   = (const float*)d_in[3];
    const float* b2   = (const float*)d_in[4];
    const int*   tids = (const int*)d_in[5];
    float* out = (float*)d_out;

    int K = in_sizes[5];
    int nwg = (K + TPW - 1) / TPW;

    const size_t emb_elems = (size_t)NUM_B * HID;
    if (ws_size >= emb_elems * sizeof(__bf16)) {
        __bf16* embbf = (__bf16*)d_ws;
        emb_to_bf16_kernel<<<(int)(emb_elems / (256 * 8)), 256, 0, stream>>>(emb, embbf);
        mtc_wmma_kernel<true><<<nwg, 256, 0, stream>>>(emb, embbf, W1, b1, W2, b2, tids, out, K);
    } else {
        mtc_wmma_kernel<false><<<nwg, 256, 0, stream>>>(emb, nullptr, W1, b1, W2, b2, tids, out, K);
    }
}
